// BilateralGridRenderer_17248588661413
// MI455X (gfx1250) — compile-verified, matrix operand order checked by reading, not run on hardware
//
#include <hip/hip_runtime.h>
#include <hip/hip_bf16.h>
#include <math.h>

typedef __attribute__((ext_vector_type(16))) _Float16 v16h;
typedef __attribute__((ext_vector_type(8)))  float    v8f;
typedef __attribute__((ext_vector_type(4)))  int      v4i;

// Problem constants (from setup_inputs): B=4, H=W=896, n_h=n_w=64
#define B_    4
#define HH    896
#define WW    896
#define NTOK  4096        // 64*64 tokens
#define MTOT  16384       // B * NTOK
#define CSEM  384
#define NA_   12
#define GS_   8
#define GL_   8
#define LS_   16
#define LL_   8
#define KMAX  384

// d_out offsets (floats): I_global, I_local, G_global, G_local, g
#define O_IG  0
#define O_IL  9633792
#define O_GG  19267584
#define O_GL  19292160
#define O_G   19390464

// workspace offsets (floats); total ~10.87M floats (~43.5 MB)
#define W_FIN   ((size_t)0)                      // fused_in [16384][384]
#define W_FUSED (W_FIN   + (size_t)MTOT*384)     // fused    [16384][256]
#define W_POOL  (W_FUSED + (size_t)MTOT*256)     // pooled   [1024][256]
#define W_LFLAT (W_POOL  + (size_t)1024*256)     // lflat    [1024][96]
#define W_FGAP  (W_LFLAT + (size_t)1024*96)      // f_gap    [4][256]
#define W_H     (W_FGAP  + (size_t)1024)         // h        [4][256]
#define W_GSEM  (W_H     + (size_t)1024)         // g_sem_patch [16384]

#if __has_builtin(__builtin_amdgcn_global_load_async_to_lds_b128) && \
    __has_builtin(__builtin_amdgcn_s_wait_asynccnt)
#define USE_ASYNC_LDS 1
#else
#define USE_ASYNC_LDS 0
#endif

union FragH { v16h v; _Float16 h[16]; };
union FragF { v8f  v; float    f[8];  };
union H8    { float4 f4; _Float16 h[8]; };

// Load 8 consecutive f32 (16B-aligned) -> f16 fragment elements [base8*8 .. +7]
__device__ __forceinline__ void ld8(const float* p, FragH& fr, int base8) {
  const float4* q = reinterpret_cast<const float4*>(p);
  float4 u = q[0], v = q[1];
  int o = base8 * 8;
  fr.h[o+0] = (_Float16)u.x; fr.h[o+1] = (_Float16)u.y;
  fr.h[o+2] = (_Float16)u.z; fr.h[o+3] = (_Float16)u.w;
  fr.h[o+4] = (_Float16)v.x; fr.h[o+5] = (_Float16)v.y;
  fr.h[o+6] = (_Float16)v.z; fr.h[o+7] = (_Float16)v.w;
}
// Load 8 consecutive f16 (16B-aligned, LDS) -> fragment elements [base8*8 .. +7]
__device__ __forceinline__ void ld8h(const _Float16* p, FragH& fr, int base8) {
  H8 t; t.f4 = *reinterpret_cast<const float4*>(p);
  int o = base8 * 8;
  #pragma unroll
  for (int j = 0; j < 8; ++j) fr.h[o+j] = t.h[j];
}

// 16B global -> LDS, async (gfx1250 ASYNCcnt path) with pass-through fallback
__device__ __forceinline__ void stage16(const float* gp, float* lp) {
#if USE_ASYNC_LDS
  // builtin expects (v4i addrspace(1)*, v4i addrspace(3)*, imm offset, imm cpol)
  __attribute__((address_space(1))) v4i* g =
      (__attribute__((address_space(1))) v4i*)
      (__attribute__((address_space(1))) void*)(const_cast<float*>(gp));
  __attribute__((address_space(3))) v4i* l =
      (__attribute__((address_space(3))) v4i*)
      (__attribute__((address_space(3))) void*)(lp);
  __builtin_amdgcn_global_load_async_to_lds_b128(g, l, 0, 0);
#else
  *reinterpret_cast<float4*>(lp) = *reinterpret_cast<const float4*>(gp);
#endif
}
__device__ __forceinline__ void stage_wait() {
#if USE_ASYNC_LDS
  __builtin_amdgcn_s_wait_asynccnt(0);
#endif
  __syncthreads();
}

// ---------------------------------------------------------------------------
// WMMA GEMM: Out[M x N] = A[M x K] * W^T (W is [N][K]) + bias.
// Block = 128 threads (4 waves). The block's 16-col weight panel is staged
// once into LDS (async copy), then shared by 4 waves, each owning one 16-row
// m-tile. grid = (N/16, ceil(M/64)). Rows >= Mvalid clamped/skipped.
// ---------------------------------------------------------------------------
__global__ void gemm16(const float* __restrict__ A, int lda,
                       const float* __restrict__ Wm, int ldw,
                       const float* __restrict__ bias,
                       float* __restrict__ Out, int ldo,
                       int Mvalid, int K, int relu) {
  __shared__ __align__(16) float Bs[16 * KMAX];   // <= 24 KB
  int tid = threadIdx.x;
  int tn  = blockIdx.x * 16;

  // stage weight panel [16][K] (f32) into LDS
  int k4 = K >> 2;
  for (int c = tid; c < 16 * k4; c += 128) {
    int row = c / k4, col4 = c % k4;
    stage16(Wm + (size_t)(tn + row) * ldw + col4 * 4, &Bs[row * K + col4 * 4]);
  }
  stage_wait();

  int wave = tid >> 5, lane = tid & 31;
  int half = lane >> 4, l16 = lane & 15;
  int tm = blockIdx.y * 64 + wave * 16;
  int ar = tm + l16; if (ar >= Mvalid) ar = Mvalid - 1;
  const float* Ap = A  + (size_t)ar * lda + half * 8;
  const float* Bp = Bs + (size_t)l16 * K + half * 8;

  FragF acc; acc.v = (v8f){0.f,0.f,0.f,0.f,0.f,0.f,0.f,0.f};
  for (int kk = 0; kk < K; kk += 32) {
    __builtin_prefetch(Ap + kk + 64, 0, 0);
    FragH a, b;
    ld8(Ap + kk,      a, 0);  ld8(Ap + kk + 16, a, 1);
    ld8(Bp + kk,      b, 0);  ld8(Bp + kk + 16, b, 1);   // LDS reads
    acc.v = __builtin_amdgcn_wmma_f32_16x16x32_f16(
        false, a.v, false, b.v, (short)0, acc.v, false, false);
  }
  float bi = bias[tn + l16];
  float* op = Out + (size_t)(tm + half * 8) * ldo + tn + l16;
  if (tm + 16 <= Mvalid) {            // full tile: branch-free stores
    #pragma unroll
    for (int r = 0; r < 8; ++r) {
      float v = acc.f[r] + bi;
      if (relu) v = fmaxf(v, 0.f);
      op[(size_t)r * ldo] = v;
    }
  } else {
    #pragma unroll
    for (int r = 0; r < 8; ++r) {
      int row = tm + r + half * 8;
      if (row < Mvalid) {
        float v = acc.f[r] + bi;
        if (relu) v = fmaxf(v, 0.f);
        Out[(size_t)row * ldo + tn + l16] = v;
      }
    }
  }
}

// ---------------------------------------------------------------------------
// Guide head: hsem = relu(F_sem @ w1^T + b1) [16384 x 64] via 4 WMMA col-tiles
// per wave; w1 staged once per block in LDS as f16 (48 KB). Then
// g_sem_patch = sigmoid(hsem . w2 + b2) via half-wave shuffle reduction.
// Block = 128 threads (4 waves, 64 rows). grid = 16384/64 = 256.
// ---------------------------------------------------------------------------
__global__ void guide_sem(const float* __restrict__ F,
                          const float* __restrict__ w1,  // [64][384]
                          const float* __restrict__ b1,  // [64]
                          const float* __restrict__ w2,  // [64]
                          const float* __restrict__ b2,  // [1]
                          float* __restrict__ outp) {    // [16384]
  __shared__ __align__(16) _Float16 Ws[64 * CSEM];       // 48 KB
  int tid = threadIdx.x;

  // stage + convert w1 -> f16 in LDS
  const int g8n = CSEM / 8;                              // 48 groups of 8 per row
  for (int c = tid; c < 64 * g8n; c += 128) {
    int row = c / g8n, g8 = c % g8n;
    const float4* gp = reinterpret_cast<const float4*>(w1 + (size_t)row * CSEM + g8 * 8);
    float4 u = gp[0], v = gp[1];
    H8 t;
    t.h[0]=(_Float16)u.x; t.h[1]=(_Float16)u.y; t.h[2]=(_Float16)u.z; t.h[3]=(_Float16)u.w;
    t.h[4]=(_Float16)v.x; t.h[5]=(_Float16)v.y; t.h[6]=(_Float16)v.z; t.h[7]=(_Float16)v.w;
    *reinterpret_cast<float4*>(&Ws[row * CSEM + g8 * 8]) = t.f4;
  }
  __syncthreads();

  int wave = tid >> 5, lane = tid & 31;
  int half = lane >> 4, l16 = lane & 15;
  int tm = blockIdx.x * 64 + wave * 16;
  const float* Ap = F + (size_t)(tm + l16) * CSEM + half * 8;

  FragF acc[4];
  #pragma unroll
  for (int t = 0; t < 4; ++t) acc[t].v = (v8f){0.f,0.f,0.f,0.f,0.f,0.f,0.f,0.f};

  for (int kk = 0; kk < CSEM; kk += 32) {
    FragH a;
    ld8(Ap + kk, a, 0);  ld8(Ap + kk + 16, a, 1);
    #pragma unroll
    for (int t = 0; t < 4; ++t) {
      const _Float16* Bp = Ws + (size_t)(t * 16 + l16) * CSEM + half * 8;
      FragH b;
      ld8h(Bp + kk, b, 0);  ld8h(Bp + kk + 16, b, 1);
      acc[t].v = __builtin_amdgcn_wmma_f32_16x16x32_f16(
          false, a.v, false, b.v, (short)0, acc[t].v, false, false);
    }
  }
  #pragma unroll
  for (int r = 0; r < 8; ++r) {
    float s = 0.f;
    #pragma unroll
    for (int t = 0; t < 4; ++t) {
      int col = t * 16 + l16;
      float h = fmaxf(acc[t].f[r] + b1[col], 0.f);
      s += h * w2[col];
    }
    for (int m = 1; m < 16; m <<= 1) s += __shfl_xor(s, m, 32);
    if (l16 == 0) {
      int row = tm + r + half * 8;
      outp[row] = 1.f / (1.f + expf(-(s + b2[0])));
    }
  }
}

// R_spatial (B,256,64,64) channel-major -> fused_in[:, 0:256] token-major
__global__ void transpose_R(const float* __restrict__ R, float* __restrict__ fin) {
  int idx = blockIdx.x * 256 + threadIdx.x;
  if (idx >= B_ * 256 * NTOK) return;
  int b = idx / (256 * NTOK);
  int r = idx % (256 * NTOK);
  int c = r / NTOK, n = r % NTOK;
  fin[((size_t)(b * NTOK + n)) * CSEM + c] = R[idx];
}

// f_gap: mean of fused over 4096 tokens -> [4][256]
__global__ void gap_k(const float* __restrict__ fused, float* __restrict__ fgap) {
  int b = blockIdx.x, c = threadIdx.x;
  const float* fp = fused + (size_t)b * NTOK * 256 + c;
  float s = 0.f;
  for (int n = 0; n < NTOK; ++n) s += fp[(size_t)n * 256];
  fgap[b * 256 + c] = s * (1.0f / NTOK);
}

// pooled: 4x4 token-block means, token-major [1024][256]
__global__ void pool_k(const float* __restrict__ fused, float* __restrict__ pooled) {
  int idx = blockIdx.x * 256 + threadIdx.x;
  if (idx >= B_ * 256 * 256) return;
  int c = idx & 255, pp = (idx >> 8) & 255, b = idx >> 16;
  int py = pp >> 4, px = pp & 15;
  float s = 0.f;
  #pragma unroll
  for (int dy = 0; dy < 4; ++dy)
    #pragma unroll
    for (int dx = 0; dx < 4; ++dx) {
      int n = (py * 4 + dy) * 64 + (px * 4 + dx);
      s += fused[((size_t)(b * NTOK + n)) * 256 + c];
    }
  pooled[idx] = s * (1.f / 16.f);
}

// lflat [1024][96] -> G_local (B,12,16,16,8) in d_out
__global__ void scatter_local(const float* __restrict__ lflat, float* __restrict__ out) {
  int idx = blockIdx.x * 256 + threadIdx.x;
  if (idx >= B_ * 256 * 96) return;
  int o = idx % 96, p = (idx / 96) % 256, b = idx / (96 * 256);
  int a = o / LL_, z = o % LL_;
  out[O_GL + (((size_t)(b * NA_ + a)) * 256 + p) * LL_ + z] = lflat[idx];
}

// ---------------------------------------------------------------------------
// Per-pixel render (Lab chroma guide + bilinear semantic guide + 2 slices).
// Grids stay in global: 122 KB/batch is L2/WGP$-resident on a 192 MB L2.
// ---------------------------------------------------------------------------
__device__ __forceinline__ float srgb_lin(float c) {
  return c > 0.04045f ? powf((c + 0.055f) * (1.f / 1.055f), 2.4f) : c * (1.f / 12.92f);
}
__device__ __forceinline__ float labf(float t) {
  return t > 0.008856f ? cbrtf(t) : 7.787f * t + 4.0f / 29.0f;
}

__device__ __forceinline__ void slice_apply(const float* __restrict__ gp, int s, int l,
                                            int x, int y, float gz,
                                            float r, float g, float b, float* o) {
  float fx = (float)x * (float)(s - 1) / (float)(WW - 1);
  float fy = (float)y * (float)(s - 1) / (float)(HH - 1);
  float fz = fminf(fmaxf(gz * (float)(l - 1), 0.f), (float)(l - 1));
  int x0 = (int)floorf(fx); int x1 = x0 + 1 < s ? x0 + 1 : s - 1;
  int y0 = (int)floorf(fy); int y1 = y0 + 1 < s ? y0 + 1 : s - 1;
  int z0 = (int)floorf(fz); int z1 = z0 + 1 < l ? z0 + 1 : l - 1;
  float wx = fx - x0, wy = fy - y0, wz = fz - z0;
  float w[8] = { (1-wx)*(1-wy)*(1-wz), wx*(1-wy)*(1-wz),
                 (1-wx)*wy*(1-wz),     wx*wy*(1-wz),
                 (1-wx)*(1-wy)*wz,     wx*(1-wy)*wz,
                 (1-wx)*wy*wz,         wx*wy*wz };
  int off[8] = { (y0*s+x0)*l+z0, (y0*s+x1)*l+z0, (y1*s+x0)*l+z0, (y1*s+x1)*l+z0,
                 (y0*s+x0)*l+z1, (y0*s+x1)*l+z1, (y1*s+x0)*l+z1, (y1*s+x1)*l+z1 };
  int ss = s * s * l;
  float coeff[12];
  #pragma unroll
  for (int c = 0; c < 12; ++c) {
    const float* cp = gp + (size_t)c * ss;
    float v = 0.f;
    #pragma unroll
    for (int j = 0; j < 8; ++j) v += w[j] * cp[off[j]];
    coeff[c] = v;
  }
  o[0] = fminf(fmaxf(coeff[0]*r + coeff[1]*g + coeff[2]*b + coeff[9],  0.f), 1.f);
  o[1] = fminf(fmaxf(coeff[3]*r + coeff[4]*g + coeff[5]*b + coeff[10], 0.f), 1.f);
  o[2] = fminf(fmaxf(coeff[6]*r + coeff[7]*g + coeff[8]*b + coeff[11], 0.f), 1.f);
}

__global__ void render_px(const float* __restrict__ img,
                          const float* __restrict__ gsem_patch,
                          float* out) {
  int idx = blockIdx.x * 256 + threadIdx.x;
  if (idx >= B_ * HH * WW) return;
  int b = idx / (HH * WW);
  int p = idx % (HH * WW);
  int y = p / WW, x = p % WW;
  const size_t pl = (size_t)HH * WW;
  const float* ip = img + (size_t)b * 3 * pl + p;
  float r = ip[0], g = ip[pl], bl = ip[2 * pl];

  float lr = srgb_lin(r), lg = srgb_lin(g), lb = srgb_lin(bl);
  float X = 0.412453f*lr + 0.35758f *lg + 0.180423f*lb;
  float Y = 0.212671f*lr + 0.71516f *lg + 0.072169f*lb;
  float Z = 0.019334f*lr + 0.119193f*lg + 0.950227f*lb;
  float fxv = labf(X * (1.f/0.950456f)), fyv = labf(Y), fzv = labf(Z * (1.f/1.088754f));
  float L = 116.f * fyv - 16.f;
  float a = 500.f * (fxv - fyv);
  float bb = 200.f * (fyv - fzv);
  float gch = fminf(fmaxf((0.5f*L + 0.25f*fabsf(a) + 0.25f*fabsf(bb)) * (1.f/114.f), 0.f), 1.f);

  float u = (x + 0.5f) * (64.f / (float)WW) - 0.5f;
  float v = (y + 0.5f) * (64.f / (float)HH) - 0.5f;
  u = fminf(fmaxf(u, 0.f), 63.f);  v = fminf(fmaxf(v, 0.f), 63.f);
  int u0 = (int)u, v0 = (int)v;
  int u1 = u0 + 1 < 64 ? u0 + 1 : 63, v1 = v0 + 1 < 64 ? v0 + 1 : 63;
  float du = u - u0, dv = v - v0;
  const float* sp = gsem_patch + (size_t)b * NTOK;
  float gs = (sp[v0*64+u0]*(1.f-du) + sp[v0*64+u1]*du) * (1.f-dv)
           + (sp[v1*64+u0]*(1.f-du) + sp[v1*64+u1]*du) * dv;

  float gg = 0.5f * gch + 0.5f * gs;
  out[O_G + (size_t)b * pl + p] = gg;

  float og[3], ol[3];
  slice_apply(out + O_GG + (size_t)b * NA_ * GS_ * GS_ * GL_, GS_, GL_, x, y, gg, r, g, bl, og);
  slice_apply(out + O_GL + (size_t)b * NA_ * LS_ * LS_ * LL_, LS_, LL_, x, y, gg, r, g, bl, ol);

  size_t o1 = (size_t)(b * 3) * pl + p;
  out[O_IG + o1         ] = og[0];
  out[O_IG + o1 +     pl] = og[1];
  out[O_IG + o1 + 2 * pl] = og[2];
  out[O_IL + o1         ] = ol[0];
  out[O_IL + o1 +     pl] = ol[1];
  out[O_IL + o1 + 2 * pl] = ol[2];
}

extern "C" void kernel_launch(void* const* d_in, const int* in_sizes, int n_in,
                              void* d_out, int out_size, void* d_ws, size_t ws_size,
                              hipStream_t stream) {
  const float* R    = (const float*)d_in[0];   // (4,256,64,64)
  const float* Fsem = (const float*)d_in[1];   // (4,4096,384)
  const float* img  = (const float*)d_in[2];   // (4,3,896,896)
  const float* dpw  = (const float*)d_in[5];   // (128,384)
  const float* dpb  = (const float*)d_in[6];
  const float* fw   = (const float*)d_in[7];   // (256,384)
  const float* fb   = (const float*)d_in[8];
  const float* g1w  = (const float*)d_in[9];   // (256,256)
  const float* g1b  = (const float*)d_in[10];
  const float* g2w  = (const float*)d_in[11];  // (6144,256)
  const float* g2b  = (const float*)d_in[12];
  const float* lw   = (const float*)d_in[13];  // (96,256)
  const float* lb   = (const float*)d_in[14];
  const float* gw1  = (const float*)d_in[15];  // (64,384)
  const float* gb1  = (const float*)d_in[16];
  const float* gw2  = (const float*)d_in[17];  // (1,64)
  const float* gb2  = (const float*)d_in[18];

  float* out = (float*)d_out;
  float* ws  = (float*)d_ws;
  float* fin    = ws + W_FIN;
  float* fused  = ws + W_FUSED;
  float* pooled = ws + W_POOL;
  float* lflat  = ws + W_LFLAT;
  float* fgap   = ws + W_FGAP;
  float* hbuf   = ws + W_H;
  float* gsem   = ws + W_GSEM;

  transpose_R<<<(B_*256*NTOK + 255) / 256, 256, 0, stream>>>(R, fin);
  // proj = F_sem @ dino_proj_w^T + b  -> columns 256:384 of fused_in
  gemm16<<<dim3(128/16, MTOT/64), 128, 0, stream>>>(Fsem, 384, dpw, 384, dpb,
                                                    fin + 256, 384, MTOT, 384, 0);
  // fused = fused_in @ fusion_w^T + b
  gemm16<<<dim3(256/16, MTOT/64), 128, 0, stream>>>(fin, 384, fw, 384, fb,
                                                    fused, 256, MTOT, 384, 0);
  // semantic guide patches
  guide_sem<<<MTOT/64, 128, 0, stream>>>(Fsem, gw1, gb1, gw2, gb2, gsem);
  // global average pool
  gap_k<<<B_, 256, 0, stream>>>(fused, fgap);
  // h = relu(f_gap @ g1_w^T + b)   (M padded 4->16, 1 block)
  gemm16<<<dim3(256/16, 1), 128, 0, stream>>>(fgap, 256, g1w, 256, g1b,
                                              hbuf, 256, B_, 256, 1);
  // G_global = h @ g2_w^T + b -> straight into d_out
  gemm16<<<dim3(6144/16, 1), 128, 0, stream>>>(hbuf, 256, g2w, 256, g2b,
                                               out + O_GG, 6144, B_, 256, 0);
  // 4x4 pooling (token-major)
  pool_k<<<(B_*256*256 + 255) / 256, 256, 0, stream>>>(fused, pooled);
  // lflat = pooled @ local_w^T + b
  gemm16<<<dim3(96/16, 1024/64), 128, 0, stream>>>(pooled, 256, lw, 256, lb,
                                                   lflat, 96, 1024, 256, 0);
  scatter_local<<<(B_*256*96 + 255) / 256, 256, 0, stream>>>(lflat, out);
  render_px<<<(B_*HH*WW + 255) / 256, 256, 0, stream>>>(img, gsem, out);
}
